// CrossTransformerBlock_28286654612087
// MI455X (gfx1250) — matrix operand order repaired and checked by reading, not compiled, not run on hardware
//
#include <hip/hip_runtime.h>

#define NQ    8192
#define NPTS  4096
#define BATCH 2
#define DIM   128
#define KNN   16
#define NB    17          // KNN + 1 global slot
#define QT    8           // queries per workgroup (fused kernel)
#define ROWS  (QT * NB)   // 136 live rows
#define ROWSP 144         // padded to 9 * 16

typedef unsigned short ushort_t;
typedef __attribute__((ext_vector_type(16))) __bf16 v16bf;
typedef __attribute__((ext_vector_type(8)))  float  v8f;

union FragBF { v16bf v; unsigned int u[8]; };

__device__ __forceinline__ ushort_t f2bf(float x) {
  union { float f; unsigned int u; } c; c.f = x;
  unsigned int u = c.u;
  u += 0x7FFFu + ((u >> 16) & 1u);              // round-to-nearest-even
  return (ushort_t)(u >> 16);
}
__device__ __forceinline__ float bf2f(ushort_t h) {
  union { unsigned int u; float f; } c; c.u = ((unsigned int)h) << 16;
  return c.f;
}

// Wave-relative LDS byte offset: generic LDS pointers carry the offset in the
// low 32 bits (aperture in the high bits), per the flat-address LDS mapping.
__device__ __forceinline__ unsigned lds_off_of(const void* p) {
  return (unsigned)(unsigned long long)p;
}

// Async DMA of one 512B row (32 lanes x 16B) global -> LDS, ASYNCcnt-tracked.
__device__ __forceinline__ void async_row_to_lds(const float* gsrc,
                                                 unsigned lds_byte_off, int lane) {
  const float* g = gsrc + lane * 4;            // 16 bytes per lane
  unsigned l = lds_byte_off + (unsigned)(lane * 16);
  asm volatile("global_load_async_to_lds_b128 %0, %1, off"
               :: "v"(l), "v"(g) : "memory");
}
__device__ __forceinline__ void wait_async0() {
  asm volatile("s_wait_asynccnt 0x0" ::: "memory");
}

// A fragment (16x32 bf16, M x K) from LDS, row-major [row][ld]
// ISA 7.12.2: lane = half*16+m holds K = (e>=4?16:0) + half*8 + 2*(e&3) (+1)
__device__ __forceinline__ v16bf load_frag_A(const ushort_t* base, int ld,
                                             int row0, int k0, int lane) {
  FragBF a;
  const int m = lane & 15, half = lane >> 4;
  const ushort_t* rp = base + (row0 + m) * ld + k0 + half * 8;
#pragma unroll
  for (int e = 0; e < 8; ++e)
    a.u[e] = *(const unsigned int*)(rp + ((e & 4) << 2) + ((e & 3) << 1));
  return a.v;
}

// B fragment (32x16 bf16, K x N) from transposed weights WT[n][k] in LDS.
// lane: n = n0 + (lane&15); VGPR e holds K = (e>=4?16:0) + half*8 + 2*(e&3) (+1)
__device__ __forceinline__ v16bf load_frag_B(const ushort_t* wt, int ld,
                                             int n0, int k0, int lane) {
  FragBF b;
  const int n = n0 + (lane & 15), half = lane >> 4;
  const ushort_t* rp = wt + n * ld + k0 + half * 8;
#pragma unroll
  for (int e = 0; e < 8; ++e)
    b.u[e] = *(const unsigned int*)(rp + ((e & 4) << 2) + ((e & 3) << 1));
  return b.v;
}

// ---------------------------------------------------------------- kNN kernel
__global__ __launch_bounds__(256) void knn_kernel(
    const float* __restrict__ xyz_q, const float* __restrict__ xyz,
    int* __restrict__ knn) {
  __shared__ float sx[1024], sy[1024];
  const int tid = threadIdx.x;
  const int b   = blockIdx.y;
  const int q   = blockIdx.x * 256 + tid;
  const float qx = xyz_q[((size_t)(b * NQ + q)) * 2 + 0];
  const float qy = xyz_q[((size_t)(b * NQ + q)) * 2 + 1];
  float bd[KNN]; int bi[KNN];
#pragma unroll
  for (int k = 0; k < KNN; ++k) { bd[k] = 3.0e38f; bi[k] = 0; }
  for (int t = 0; t < NPTS; t += 1024) {
    for (int i = tid; i < 1024; i += 256) {
      sx[i] = xyz[((size_t)(b * NPTS + t + i)) * 3 + 0];
      sy[i] = xyz[((size_t)(b * NPTS + t + i)) * 3 + 1];
    }
    if (t + 1024 < NPTS)                        // L2 prefetch of next tile
      __builtin_prefetch(&xyz[((size_t)(b * NPTS + t + 1024 + tid * 4)) * 3], 0, 3);
    __syncthreads();
#pragma unroll 4
    for (int j = 0; j < 1024; ++j) {
      const float dx = qx - sx[j], dy = qy - sy[j];
      const float d  = dx * dx + dy * dy;
      if (d < bd[KNN - 1]) {                    // quick reject keeps VALU cheap
        bd[KNN - 1] = d; bi[KNN - 1] = t + j;
#pragma unroll
        for (int s = KNN - 1; s > 0; --s) {
          if (bd[s] < bd[s - 1]) {
            float td = bd[s]; bd[s] = bd[s - 1]; bd[s - 1] = td;
            int   ti = bi[s]; bi[s] = bi[s - 1]; bi[s - 1] = ti;
          }
        }
      }
    }
    __syncthreads();
  }
#pragma unroll
  for (int k = 0; k < KNN; ++k)
    knn[((size_t)(b * NQ + q)) * KNN + k] = bi[k];
}

// ------------------------------------------------- K/V projection (WMMA GEMM)
__global__ __launch_bounds__(256) void kv_project_kernel(
    const float* __restrict__ points, const float* __restrict__ wk,
    const float* __restrict__ wv, float* __restrict__ Kp,
    float* __restrict__ Vp) {
  __shared__ ushort_t sA[DIM * DIM];
  __shared__ ushort_t sW[DIM * DIM];
  const int tid = threadIdx.x, wave = tid >> 5, lane = tid & 31;
  const size_t row0 = (size_t)blockIdx.x * DIM;   // flat over B*N rows

  for (int i = tid; i < DIM * DIM; i += 256)
    sA[i] = f2bf(points[row0 * DIM + i]);
  for (int i = tid; i < DIM * DIM; i += 256) {
    const int n = i >> 7, k = i & 127;
    sW[i] = f2bf(wk[k * DIM + n]);                // W^T layout [n][k]
  }
  __syncthreads();
  {
    const int mt = wave;
#pragma unroll
    for (int nt = 0; nt < 8; ++nt) {
      v8f acc = {0.f, 0.f, 0.f, 0.f, 0.f, 0.f, 0.f, 0.f};
#pragma unroll
      for (int kt = 0; kt < 4; ++kt) {
        v16bf a = load_frag_A(sA, DIM, mt * 16, kt * 32, lane);
        v16bf w = load_frag_B(sW, DIM, nt * 16, kt * 32, lane);
        acc = __builtin_amdgcn_wmma_f32_16x16x32_bf16(false, a, false, w,
                                                      (short)0, acc, false, false);
      }
      const int n = nt * 16 + (lane & 15), half = lane >> 4;
#pragma unroll
      for (int d = 0; d < 8; ++d)
        Kp[(row0 + mt * 16 + d + half * 8) * DIM + n] = acc[d];
    }
  }
  __syncthreads();
  for (int i = tid; i < DIM * DIM; i += 256) {
    const int n = i >> 7, k = i & 127;
    sW[i] = f2bf(wv[k * DIM + n]);
  }
  __syncthreads();
  {
    const int mt = wave;
#pragma unroll
    for (int nt = 0; nt < 8; ++nt) {
      v8f acc = {0.f, 0.f, 0.f, 0.f, 0.f, 0.f, 0.f, 0.f};
#pragma unroll
      for (int kt = 0; kt < 4; ++kt) {
        v16bf a = load_frag_A(sA, DIM, mt * 16, kt * 32, lane);
        v16bf w = load_frag_B(sW, DIM, nt * 16, kt * 32, lane);
        acc = __builtin_amdgcn_wmma_f32_16x16x32_bf16(false, a, false, w,
                                                      (short)0, acc, false, false);
      }
      const int n = nt * 16 + (lane & 15), half = lane >> 4;
#pragma unroll
      for (int d = 0; d < 8; ++d)
        Vp[(row0 + mt * 16 + d + half * 8) * DIM + n] = acc[d];
    }
  }
}

// ----------------------------------------------------- global latent vectors
__global__ __launch_bounds__(128) void globals_kernel(
    const float* __restrict__ lat, const float* __restrict__ wq,
    const float* __restrict__ wkg, const float* __restrict__ wvg,
    float* __restrict__ qg, float* __restrict__ kg, float* __restrict__ vg) {
  const int b = blockIdx.x, f = threadIdx.x;
  float aq = 0.f, ak = 0.f, av = 0.f;
  for (int k = 0; k < DIM; ++k) {
    const float l = lat[b * DIM + k];
    aq += l * wq[k * DIM + f];
    ak += l * wkg[k * DIM + f];
    av += l * wvg[k * DIM + f];
  }
  qg[b * DIM + f] = aq; kg[b * DIM + f] = ak; vg[b * DIM + f] = av;
}

// -------------------------------------- fused pos-enc + MLP + softmax kernel
__global__ __launch_bounds__(256) void fused_attn_kernel(
    const float* __restrict__ xyz_q, const float* __restrict__ xyz,
    const int* __restrict__ knn, const float* __restrict__ Kp,
    const float* __restrict__ Vp, const float* __restrict__ qg,
    const float* __restrict__ kgv, const float* __restrict__ vgv,
    const float* __restrict__ g1, const float* __restrict__ b1,
    const float* __restrict__ g2, const float* __restrict__ b2,
    const float* __restrict__ pe_w, const float* __restrict__ pe_b,
    float* __restrict__ out) {
  __shared__ ushort_t sm_wbuf[DIM * DIM];       // 32 KB, reused per stage
  __shared__ ushort_t sm_pos[ROWSP * DIM];      // 36 KB bf16 pos_encode
  __shared__ ushort_t sm_h[ROWSP * DIM];        // 36 KB h, later logits
  __shared__ ushort_t sm_t[ROWSP * DIM];        // 36 KB; aliases pemb early
  __shared__ float    sm_kv[128 * DIM];         // 64 KB async K/V row staging
  __shared__ float sm_qvec[DIM], sm_kg[DIM], sm_vg[DIM];
  __shared__ float sm_b1[DIM], sm_b2[DIM], sm_peb[DIM];
  __shared__ int sm_idx[QT * KNN];
  ushort_t* sm_pemb = sm_t;                     // [128][64], dead before GEMM1

  const int tid = threadIdx.x;
  const int wave = tid >> 5, lane = tid & 31;
  const int b  = blockIdx.y;
  const int q0 = blockIdx.x * QT;

  // ---- stage 0a: indices + small vectors into LDS
  if (tid < DIM) {
    sm_qvec[tid] = qg[b * DIM + tid];
    sm_kg[tid]   = kgv[b * DIM + tid];
    sm_vg[tid]   = vgv[b * DIM + tid];
    sm_b1[tid]   = b1[tid];
    sm_b2[tid]   = b2[tid];
    sm_peb[tid]  = pe_b[tid];
    sm_idx[tid]  = knn[((size_t)(b * NQ + q0 + (tid >> 4))) * KNN + (tid & 15)];
  }
  __syncthreads();

  // ---- stage 0b: kick off async Kp row gather (overlaps stage 2 WMMA)
  const unsigned kv_base = lds_off_of(sm_kv);
  for (int r = wave; r < 128; r += 8) {         // row r = n*QT + q, n < 16
    const int n = r >> 3, q = r & 7;
    const float* g = Kp + ((size_t)(b * NPTS + sm_idx[q * KNN + n])) * DIM;
    async_row_to_lds(g, kv_base + (unsigned)(r * 512), lane);
  }
  // pos_embed rows (r = n*QT + q), zero pads, load pe_w^T (33 -> 64 pad)
  for (int i = tid; i < 16 * DIM; i += 256) sm_pos[128 * DIM + i] = 0; // slot16+pad
  for (int i = tid; i < 8 * DIM; i += 256)  sm_h[136 * DIM + i] = 0;  // h pad rows
  for (int i = tid; i < DIM * 64; i += 256) {
    const int n = i >> 6, k = i & 63;
    sm_wbuf[i] = (k < 33) ? f2bf(pe_w[k * DIM + n]) : (ushort_t)0;
  }
  if (tid < QT * KNN) {
    const int q = tid >> 4, n = tid & 15;
    const int prow = n * QT + q;
    const float qx = xyz_q[((size_t)(b * NQ + q0 + q)) * 2 + 0];
    const float qy = xyz_q[((size_t)(b * NQ + q0 + q)) * 2 + 1];
    const int id = sm_idx[tid];
    const float px = xyz[((size_t)(b * NPTS + id)) * 3 + 0];
    const float py = xyz[((size_t)(b * NPTS + id)) * 3 + 1];
    const float pz = xyz[((size_t)(b * NPTS + id)) * 3 + 2];
    float pd[3] = {qx - px, qy - py, pz};
    ushort_t* row = sm_pemb + prow * 64;
    row[0] = f2bf(pd[0]); row[1] = f2bf(pd[1]); row[2] = f2bf(pd[2]);
    int c = 3;
#pragma unroll
    for (int fi = 0; fi < 5; ++fi) {
      const float fr = 1.0f + 7.75f * fi;       // linspace(1, 32, 5)
#pragma unroll
      for (int a = 0; a < 3; ++a) {
        row[c + a]     = f2bf(__sinf(pd[a] * fr));
        row[c + 3 + a] = f2bf(__cosf(pd[a] * fr));
      }
      c += 6;
    }
#pragma unroll
    for (int a2 = 33; a2 < 64; ++a2) row[a2] = 0;
  }
  __syncthreads();

  // ---- stage 2: pos_encode = pos_embed @ pe_w + pe_b   (WMMA, K padded to 64)
  {
    const int mt = wave;                        // 8 M-tiles, one per wave
#pragma unroll
    for (int nt = 0; nt < 8; ++nt) {
      v8f acc = {0.f, 0.f, 0.f, 0.f, 0.f, 0.f, 0.f, 0.f};
#pragma unroll
      for (int ks = 0; ks < 2; ++ks) {
        v16bf a = load_frag_A(sm_pemb, 64, mt * 16, ks * 32, lane);
        v16bf w = load_frag_B(sm_wbuf, 64, nt * 16, ks * 32, lane);
        acc = __builtin_amdgcn_wmma_f32_16x16x32_bf16(false, a, false, w,
                                                      (short)0, acc, false, false);
      }
      const int n = nt * 16 + (lane & 15), half = lane >> 4;
      const float bias = sm_peb[n];
#pragma unroll
      for (int d = 0; d < 8; ++d)
        sm_pos[(mt * 16 + d + half * 8) * DIM + n] = f2bf(acc[d] + bias);
    }
  }
  wait_async0();                                // Kp rows landed in sm_kv
  __syncthreads();

  // ---- stage 3: h = q_attn - k_attn + pos_encode (K rows now in LDS); load g1^T
  for (int i = tid; i < 128 * DIM; i += 256)    // neighbor rows
    sm_h[i] = f2bf(sm_qvec[i & 127] - sm_kv[i] + bf2f(sm_pos[i]));
  for (int i = tid; i < 8 * DIM; i += 256) {    // global-slot rows (pos == 0)
    const int cix = i & 127;
    sm_h[128 * DIM + i] = f2bf(sm_qvec[cix] - sm_kg[cix]);
  }
  for (int i = tid; i < DIM * DIM; i += 256) {
    const int n = i >> 7, k = i & 127;
    sm_wbuf[i] = f2bf(g1[k * DIM + n]);
  }
  __syncthreads();

  // ---- kick off async Vp row gather (overlaps GEMM1 + GEMM2)
  for (int r = wave; r < 128; r += 8) {
    const int n = r >> 3, q = r & 7;
    const float* g = Vp + ((size_t)(b * NPTS + sm_idx[q * KNN + n])) * DIM;
    async_row_to_lds(g, kv_base + (unsigned)(r * 512), lane);
  }

  // ---- stage 4: t = relu(h @ gamma_w1 + b1)
  for (int mt = wave; mt < 9; mt += 8) {
#pragma unroll
    for (int nt = 0; nt < 8; ++nt) {
      v8f acc = {0.f, 0.f, 0.f, 0.f, 0.f, 0.f, 0.f, 0.f};
#pragma unroll
      for (int kt = 0; kt < 4; ++kt) {
        v16bf a = load_frag_A(sm_h, DIM, mt * 16, kt * 32, lane);
        v16bf w = load_frag_B(sm_wbuf, DIM, nt * 16, kt * 32, lane);
        acc = __builtin_amdgcn_wmma_f32_16x16x32_bf16(false, a, false, w,
                                                      (short)0, acc, false, false);
      }
      const int n = nt * 16 + (lane & 15), half = lane >> 4;
      const float bias = sm_b1[n];
#pragma unroll
      for (int d = 0; d < 8; ++d) {
        const float vv = acc[d] + bias;
        sm_t[(mt * 16 + d + half * 8) * DIM + n] = f2bf(vv > 0.f ? vv : 0.f);
      }
    }
  }
  __syncthreads();
  for (int i = tid; i < DIM * DIM; i += 256) {
    const int n = i >> 7, k = i & 127;
    sm_wbuf[i] = f2bf(g2[k * DIM + n]);
  }
  __syncthreads();

  // ---- stage 5: logits = t @ gamma_w2 + b2  (written over sm_h)
  for (int mt = wave; mt < 9; mt += 8) {
#pragma unroll
    for (int nt = 0; nt < 8; ++nt) {
      v8f acc = {0.f, 0.f, 0.f, 0.f, 0.f, 0.f, 0.f, 0.f};
#pragma unroll
      for (int kt = 0; kt < 4; ++kt) {
        v16bf a = load_frag_A(sm_t, DIM, mt * 16, kt * 32, lane);
        v16bf w = load_frag_B(sm_wbuf, DIM, nt * 16, kt * 32, lane);
        acc = __builtin_amdgcn_wmma_f32_16x16x32_bf16(false, a, false, w,
                                                      (short)0, acc, false, false);
      }
      const int n = nt * 16 + (lane & 15), half = lane >> 4;
      const float bias = sm_b2[n];
#pragma unroll
      for (int d = 0; d < 8; ++d)
        sm_h[(mt * 16 + d + half * 8) * DIM + n] = f2bf(acc[d] + bias);
    }
  }
  wait_async0();                                // Vp rows landed in sm_kv
  __syncthreads();

  // ---- stage 6: softmax over 17 neighbor slots + weighted sum of (v + pos)
  for (int i = tid; i < QT * DIM; i += 256) {
    const int q = i >> 7, f = i & 127;
    float l[NB];
    float lmax = -3.0e38f;
#pragma unroll
    for (int n = 0; n < NB; ++n) {
      l[n] = bf2f(sm_h[(n * QT + q) * DIM + f]);
      lmax = fmaxf(lmax, l[n]);
    }
    float s = 0.f;
#pragma unroll
    for (int n = 0; n < NB; ++n) { l[n] = __expf(l[n] - lmax); s += l[n]; }
    const float inv = 1.f / s;
    float acc = 0.f;
#pragma unroll
    for (int n = 0; n < KNN; ++n)
      acc += l[n] * (sm_kv[(n * QT + q) * DIM + f] +
                     bf2f(sm_pos[(n * QT + q) * DIM + f]));
    acc += l[KNN] * sm_vg[f];                   // global slot: pos == 0
    out[((size_t)(b * NQ + q0 + q)) * DIM + f] = acc * inv;
  }
}

// ---------------------------------------------------------------------------
extern "C" void kernel_launch(void* const* d_in, const int* in_sizes, int n_in,
                              void* d_out, int out_size, void* d_ws,
                              size_t ws_size, hipStream_t stream) {
  (void)in_sizes; (void)n_in; (void)out_size; (void)ws_size;
  const float* xyz_q  = (const float*)d_in[0];
  const float* lat    = (const float*)d_in[1];
  const float* xyz    = (const float*)d_in[2];
  const float* points = (const float*)d_in[3];
  const float* w_qs   = (const float*)d_in[4];
  const float* w_ks   = (const float*)d_in[5];
  const float* w_vs   = (const float*)d_in[6];
  const float* w_kg   = (const float*)d_in[7];
  const float* w_vg   = (const float*)d_in[8];
  const float* g1     = (const float*)d_in[9];
  const float* b1     = (const float*)d_in[10];
  const float* g2     = (const float*)d_in[11];
  const float* b2     = (const float*)d_in[12];
  const float* pe_w   = (const float*)d_in[13];
  const float* pe_b   = (const float*)d_in[14];

  char* ws = (char*)d_ws;
  int*   knn = (int*)ws;   ws += (size_t)BATCH * NQ * KNN * sizeof(int);
  float* Kp  = (float*)ws; ws += (size_t)BATCH * NPTS * DIM * sizeof(float);
  float* Vp  = (float*)ws; ws += (size_t)BATCH * NPTS * DIM * sizeof(float);
  float* qg  = (float*)ws; ws += (size_t)BATCH * DIM * sizeof(float);
  float* kg  = (float*)ws; ws += (size_t)BATCH * DIM * sizeof(float);
  float* vg  = (float*)ws;

  knn_kernel<<<dim3(NQ / 256, BATCH), 256, 0, stream>>>(xyz_q, xyz, knn);
  kv_project_kernel<<<dim3((BATCH * NPTS) / DIM), 256, 0, stream>>>(points, w_ks,
                                                                    w_vs, Kp, Vp);
  globals_kernel<<<dim3(BATCH), 128, 0, stream>>>(lat, w_qs, w_kg, w_vg, qg, kg, vg);
  fused_attn_kernel<<<dim3(NQ / QT, BATCH), 256, 0, stream>>>(
      xyz_q, xyz, knn, Kp, Vp, qg, kg, vg, g1, b1, g2, b2, pe_w, pe_b,
      (float*)d_out);
}
